// SwinFusionLayer_790273982556
// MI455X (gfx1250) — matrix operand order repaired and checked by reading, not compiled, not run on hardware
//
#include <hip/hip_runtime.h>

// ---------------- constants ----------------
static constexpr int BF   = 16;        // B*T
static constexpr int C    = 128;
static constexpr int HH   = 126;
static constexpr int WW   = 126;
static constexpr int HW   = HH * WW;   // 15876
static constexpr int M    = BF * HW;   // 254016 tokens
static constexpr int NHH  = 4;
static constexpr int NWIN = 42;        // windows per side
static constexpr int NWPI = NWIN * NWIN; // 1764 windows per image
static constexpr int TOTW = BF * NWPI;   // 28224 windows total
static constexpr int MT   = M / 16;    // 15876 M-tiles

typedef __bf16 bf16x16 __attribute__((ext_vector_type(16)));
typedef __bf16 bf16x8  __attribute__((ext_vector_type(8)));
typedef float  f32x8   __attribute__((ext_vector_type(8)));

// ---------------- helpers ----------------
__device__ __forceinline__ unsigned short f2bf(float f) {
    union { float f; unsigned u; } v; v.f = f;
    unsigned u = v.u + 0x7FFFu + ((v.u >> 16) & 1u);
    return (unsigned short)(u >> 16);
}
__device__ __forceinline__ float bf2f(unsigned short h) {
    union { unsigned u; float f; } v; v.u = ((unsigned)h) << 16;
    return v.f;
}

// gfx1250 async global->LDS copy (ASYNCcnt path, ISA 15.18 opcode 98)
__device__ __forceinline__ void async_g2l_b128(unsigned lds_byte_addr,
                                               unsigned long long gaddr) {
    asm volatile("global_load_async_to_lds_b128 %0, %1, off"
                 :: "v"(lds_byte_addr), "v"(gaddr) : "memory");
}
__device__ __forceinline__ void wait_asynccnt0() {
    asm volatile("s_wait_asynccnt 0x0" ::: "memory");
}

// ---------------- f32 -> bf16 convert (weights) ----------------
__global__ void cvt_bf16_kernel(const float* __restrict__ in,
                                unsigned short* __restrict__ out, int n) {
    int i = blockIdx.x * 256 + threadIdx.x;
    if (i < n) out[i] = f2bf(in[i]);
}

// ---------------- LayerNorm over C, input in (BF, C, HW) layout ----------------
__global__ __launch_bounds__(256) void ln1_kernel(const float* __restrict__ x,
                                                  const float* __restrict__ w,
                                                  const float* __restrict__ b,
                                                  unsigned short* __restrict__ out) {
    const int wave = threadIdx.x >> 5, lane = threadIdx.x & 31;
    const int t = blockIdx.x * 8 + wave;           // always < M (M % 8 == 0)
    const int bfi = t / HW, p = t - bfi * HW;
    const float* base = x + (size_t)bfi * C * HW + p;
    float v[4]; float s = 0.f;
#pragma unroll
    for (int i = 0; i < 4; ++i) { v[i] = base[(size_t)(lane + 32 * i) * HW]; s += v[i]; }
    for (int o = 16; o; o >>= 1) s += __shfl_xor(s, o, 32);
    const float mean = s * (1.f / 128.f);
    float q = 0.f;
#pragma unroll
    for (int i = 0; i < 4; ++i) { float d = v[i] - mean; q += d * d; }
    for (int o = 16; o; o >>= 1) q += __shfl_xor(q, o, 32);
    const float rstd = rsqrtf(q * (1.f / 128.f) + 1e-5f);
#pragma unroll
    for (int i = 0; i < 4; ++i) {
        int c = lane + 32 * i;
        out[(size_t)t * C + c] = f2bf((v[i] - mean) * rstd * w[c] + b[c]);
    }
}

// ---------------- LayerNorm over C, input token-major f32 ----------------
__global__ __launch_bounds__(256) void ln2_kernel(const float* __restrict__ xo,
                                                  const float* __restrict__ w,
                                                  const float* __restrict__ b,
                                                  unsigned short* __restrict__ out) {
    const int wave = threadIdx.x >> 5, lane = threadIdx.x & 31;
    const int t = blockIdx.x * 8 + wave;
    const float* base = xo + (size_t)t * C;
    float v[4]; float s = 0.f;
#pragma unroll
    for (int i = 0; i < 4; ++i) { v[i] = base[lane + 32 * i]; s += v[i]; }
    for (int o = 16; o; o >>= 1) s += __shfl_xor(s, o, 32);
    const float mean = s * (1.f / 128.f);
    float q = 0.f;
#pragma unroll
    for (int i = 0; i < 4; ++i) { float d = v[i] - mean; q += d * d; }
    for (int o = 16; o; o >>= 1) q += __shfl_xor(q, o, 32);
    const float rstd = rsqrtf(q * (1.f / 128.f) + 1e-5f);
#pragma unroll
    for (int i = 0; i < 4; ++i) {
        int c = lane + 32 * i;
        out[(size_t)t * C + c] = f2bf((v[i] - mean) * rstd * w[c] + b[c]);
    }
}

// ---------------- bf16 WMMA GEMM over a 128-column chunk, 2 M-tiles per wave ----
// out[:, colBase : colBase+128] = A[M,128] @ W[colBase:colBase+128, 128]^T + bias
// W pointer passed pre-offset to the chunk (128x128 bf16 = 32 KB staged in LDS
// via gfx1250 async-to-LDS). Each B fragment read from LDS feeds 2 WMMAs.
// EPI: 0 = store bf16 (ld = LDN), 1 = exact GELU -> bf16,
//      2 = + residual from (BF,C,HW) layout -> f32 token-major (proj)
//      3 = + residual f32 token-major -> f32 (BF,C,HW) layout (fc2 -> d_out)
template <int LDN, int EPI>
__global__ __launch_bounds__(256) void gemm2_kernel(const unsigned short* __restrict__ A,
                                                    const unsigned short* __restrict__ W,
                                                    const float* __restrict__ bias,
                                                    unsigned short* __restrict__ outB,
                                                    float* __restrict__ outF,
                                                    const float* __restrict__ residT,
                                                    const float* __restrict__ residF,
                                                    int colBase) {
    __shared__ __align__(16) unsigned short sW[128 * 128];   // 32 KB weight chunk

    const int tid  = threadIdx.x;
    const int wave = tid >> 5, lane = tid & 31;

    // ---- async-stage the weight chunk into LDS (gfx1250 ASYNCcnt path) ----
    {
        const unsigned sbase = (unsigned)(size_t)&sW[0];
        const unsigned long long g = (unsigned long long)(const void*)W;
#pragma unroll
        for (int it = 0; it < 8; ++it) {
            const unsigned off = (unsigned)(it * 256 + tid) * 16u;  // 16B chunks
            async_g2l_b128(sbase + off, g + off);
        }
        wait_asynccnt0();
    }
    __syncthreads();

    const int mt0 = blockIdx.x * 16 + wave * 2;
    const int mt1 = mt0 + 1;
    const bool v0 = (mt0 < MT), v1 = (mt1 < MT);
    const int row0 = (v0 ? mt0 : 0) * 16;          // clamp: safe loads, EXEC stays full
    const int row1 = (v1 ? mt1 : 0) * 16;
    const int lh = lane >> 4;
    const int lr = lane & 15;

    // A fragments (16x32 bf16, ISA layout): lanes 0-15 K [0..7],[16..23];
    // lanes 16-31 K [8..15],[24..31]; row M = lr.
    const __bf16* Ab = (const __bf16*)A;
    bf16x16 afr0[4], afr1[4];
#pragma unroll
    for (int ks = 0; ks < 4; ++ks) {
        const __bf16* p0 = Ab + (size_t)(row0 + lr) * 128 + ks * 32 + lh * 8;
        bf16x8 lo0 = *(const bf16x8*)p0;
        bf16x8 hi0 = *(const bf16x8*)(p0 + 16);
        afr0[ks] = __builtin_shufflevector(lo0, hi0, 0, 1, 2, 3, 4, 5, 6, 7,
                                           8, 9, 10, 11, 12, 13, 14, 15);
        const __bf16* p1 = Ab + (size_t)(row1 + lr) * 128 + ks * 32 + lh * 8;
        bf16x8 lo1 = *(const bf16x8*)p1;
        bf16x8 hi1 = *(const bf16x8*)(p1 + 16);
        afr1[ks] = __builtin_shufflevector(lo1, hi1, 0, 1, 2, 3, 4, 5, 6, 7,
                                           8, 9, 10, 11, 12, 13, 14, 15);
    }

    const __bf16* Ws = (const __bf16*)&sW[0];
#pragma unroll
    for (int nt = 0; nt < 8; ++nt) {
        f32x8 acc0 = {0.f, 0.f, 0.f, 0.f, 0.f, 0.f, 0.f, 0.f};
        f32x8 acc1 = {0.f, 0.f, 0.f, 0.f, 0.f, 0.f, 0.f, 0.f};
#pragma unroll
        for (int ks = 0; ks < 4; ++ks) {
            // B fragment (32x16): lanes 0-15 column n, K [0..7],[16..23]; lanes 16-31 K [8..15],[24..31]
            const __bf16* pb = Ws + (nt * 16 + lr) * 128 + ks * 32 + lh * 8;
            bf16x8 blo = *(const bf16x8*)pb;
            bf16x8 bhi = *(const bf16x8*)(pb + 16);
            bf16x16 bfr = __builtin_shufflevector(blo, bhi, 0, 1, 2, 3, 4, 5, 6, 7,
                                                  8, 9, 10, 11, 12, 13, 14, 15);
            acc0 = __builtin_amdgcn_wmma_f32_16x16x32_bf16(
                false, afr0[ks], false, bfr, (short)0, acc0, false, false);
            acc1 = __builtin_amdgcn_wmma_f32_16x16x32_bf16(
                false, afr1[ks], false, bfr, (short)0, acc1, false, false);
        }
        const int col = colBase + nt * 16 + lr;
        const float bv = bias[col];
#pragma unroll
        for (int t2 = 0; t2 < 2; ++t2) {
            if (t2 == 0 ? !v0 : !v1) continue;     // wave-uniform
            const f32x8& acc = (t2 == 0) ? acc0 : acc1;
            const int rbase = (t2 == 0 ? row0 : row1) + lh * 8;
#pragma unroll
            for (int j = 0; j < 8; ++j) {
                const int m = rbase + j;
                float v = acc[j] + bv;
                if constexpr (EPI == 0) {
                    outB[(size_t)m * LDN + col] = f2bf(v);
                } else if constexpr (EPI == 1) {
                    float g = 0.5f * v * (1.f + erff(v * 0.70710678118654752f));
                    outB[(size_t)m * LDN + col] = f2bf(g);
                } else if constexpr (EPI == 2) {
                    const int bfi = m / HW, p = m - bfi * HW;
                    v += residT[(size_t)bfi * C * HW + (size_t)col * HW + p];
                    outF[(size_t)m * C + col] = v;
                } else {
                    v += residF[(size_t)m * C + col];
                    const int bfi = m / HW, p = m - bfi * HW;
                    outF[(size_t)bfi * C * HW + (size_t)col * HW + p] = v;
                }
            }
        }
    }
}

// ---------------- shifted-window attention, one wave per 3x3 window ----------------
__global__ __launch_bounds__(256) void attn_kernel(const unsigned short* __restrict__ qkv,
                                                   const float* __restrict__ relb,
                                                   unsigned short* __restrict__ out) {
    __shared__ __align__(16) unsigned short sQ[8][3456];  // 9 tokens * 384 halfs / window
    __shared__ float sP[8][81];
    const int wave = threadIdx.x >> 5, lane = threadIdx.x & 31;
    const int w = blockIdx.x * 8 + wave;           // TOTW % 8 == 0, always valid
    const int bfi = w / NWPI;
    const int rem = w - bfi * NWPI;
    const int wy = rem / NWIN, wx = rem - wy * NWIN;

    // async-stage q,k,v for 9 tokens (cyclic un-shift gather: orig = (rolled+1) mod 126)
    {
        const unsigned sbase = (unsigned)(size_t)&sQ[wave][0];
        const unsigned long long qb = (unsigned long long)(const void*)qkv;
        for (int tk = 0; tk < 9; ++tk) {
            const int i = tk / 3, j = tk - 3 * (tk / 3);
            int r = wy * 3 + i + 1; if (r >= HH) r -= HH;
            int c = wx * 3 + j + 1; if (c >= WW) c -= WW;
            const unsigned long long tok = (unsigned long long)bfi * HW +
                                           (unsigned long long)r * WW + c;
            const unsigned long long src = qb + tok * 768ull;   // 384 halfs = 768 B
            const unsigned dstb = sbase + (unsigned)tk * 768u;  // 48 x 16B chunks
            async_g2l_b128(dstb + lane * 16u, src + lane * 16u);
            if (lane < 16)
                async_g2l_b128(dstb + (lane + 32u) * 16u, src + (lane + 32u) * 16u);
        }
        wait_asynccnt0();
    }
    __syncthreads();

    const unsigned short* w16 = &sQ[wave][0];
    const float scale = 0.1767766952966369f;       // 32^-0.5

    for (int h = 0; h < NHH; ++h) {
        if (lane < 9) {
            const int n = lane, i = n / 3, j = n - 3 * (n / 3);
            const int rr = wy * 3 + i, cc = wx * 3 + j;
            const int regn = ((rr < 123) ? 0 : (rr < 125 ? 1 : 2)) * 3 +
                             ((cc < 123) ? 0 : (cc < 125 ? 1 : 2));
            float s[9]; float mx = -1e30f;
            for (int m = 0; m < 9; ++m) {
                const int i2 = m / 3, j2 = m - 3 * (m / 3);
                const int rr2 = wy * 3 + i2, cc2 = wx * 3 + j2;
                const int regm = ((rr2 < 123) ? 0 : (rr2 < 125 ? 1 : 2)) * 3 +
                                 ((cc2 < 123) ? 0 : (cc2 < 125 ? 1 : 2));
                float dot = 0.f;
#pragma unroll
                for (int d = 0; d < 32; ++d)
                    dot += bf2f(w16[n * 384 + h * 32 + d]) *
                           bf2f(w16[m * 384 + 128 + h * 32 + d]);
                const int ridx = (i - i2 + 2) * 5 + (j - j2 + 2);
                float sc = dot * scale + relb[ridx * NHH + h] +
                           ((regm != regn) ? -100.f : 0.f);
                s[m] = sc; mx = fmaxf(mx, sc);
            }
            float den = 0.f;
            for (int m = 0; m < 9; ++m) { s[m] = __expf(s[m] - mx); den += s[m]; }
            const float inv = 1.f / den;
            for (int m = 0; m < 9; ++m) sP[wave][n * 9 + m] = s[m] * inv;
        }
        __syncthreads();
        for (int n = 0; n < 9; ++n) {
            float val = 0.f;
#pragma unroll
            for (int m = 0; m < 9; ++m)
                val += sP[wave][n * 9 + m] * bf2f(w16[m * 384 + 256 + h * 32 + lane]);
            const int i = n / 3, j = n - 3 * (n / 3);
            int r = wy * 3 + i + 1; if (r >= HH) r -= HH;
            int c = wx * 3 + j + 1; if (c >= WW) c -= WW;
            const size_t tok = (size_t)bfi * HW + (size_t)r * WW + c;
            out[tok * C + h * 32 + lane] = f2bf(val);
        }
        __syncthreads();
    }
}

// ---------------- launch ----------------
extern "C" void kernel_launch(void* const* d_in, const int* in_sizes, int n_in,
                              void* d_out, int out_size, void* d_ws, size_t ws_size,
                              hipStream_t stream) {
    const float* x       = (const float*)d_in[0];
    const float* qkv_w   = (const float*)d_in[1];
    const float* qkv_b   = (const float*)d_in[2];
    const float* proj_w  = (const float*)d_in[3];
    const float* proj_b  = (const float*)d_in[4];
    const float* rel_b   = (const float*)d_in[5];
    const float* n1_w    = (const float*)d_in[6];
    const float* n1_b    = (const float*)d_in[7];
    const float* n2_w    = (const float*)d_in[8];
    const float* n2_b    = (const float*)d_in[9];
    const float* fc1_w   = (const float*)d_in[10];
    const float* fc1_b   = (const float*)d_in[11];
    const float* fc2_w   = (const float*)d_in[12];
    const float* fc2_b   = (const float*)d_in[13];
    float* out = (float*)d_out;

    // scratch layout
    char* ws = (char*)d_ws;
    unsigned short* wq  = (unsigned short*)(ws + 0);          // 384*128 bf16
    unsigned short* wp  = (unsigned short*)(ws + 98304);      // 128*128 bf16
    unsigned short* w1  = (unsigned short*)(ws + 131072);
    unsigned short* w2  = (unsigned short*)(ws + 163840);
    unsigned short* buf0 = (unsigned short*)(ws + 196608);    // M*C bf16
    unsigned short* buf1 = (unsigned short*)(ws + 196608 + (size_t)M * C * 2);        // M*384 bf16
    float*          buf2 = (float*)(ws + 196608 + (size_t)M * C * 2 + (size_t)M * 384 * 2); // M*C f32

    const int GB = (MT + 15) / 16;   // gemm blocks (16 M-tiles per block)

    // 1. convert weights to bf16
    cvt_bf16_kernel<<<(384 * 128 + 255) / 256, 256, 0, stream>>>(qkv_w, wq, 384 * 128);
    cvt_bf16_kernel<<<(128 * 128 + 255) / 256, 256, 0, stream>>>(proj_w, wp, 128 * 128);
    cvt_bf16_kernel<<<(128 * 128 + 255) / 256, 256, 0, stream>>>(fc1_w, w1, 128 * 128);
    cvt_bf16_kernel<<<(128 * 128 + 255) / 256, 256, 0, stream>>>(fc2_w, w2, 128 * 128);

    // 2. LN1 -> bf16
    ln1_kernel<<<M / 8, 256, 0, stream>>>(x, n1_w, n1_b, buf0);

    // 3. QKV GEMM: three 128-column chunks
    gemm2_kernel<384, 0><<<GB, 256, 0, stream>>>(buf0, wq +   0 * 128, qkv_b, buf1,
                                                 nullptr, nullptr, nullptr, 0);
    gemm2_kernel<384, 0><<<GB, 256, 0, stream>>>(buf0, wq + 128 * 128, qkv_b, buf1,
                                                 nullptr, nullptr, nullptr, 128);
    gemm2_kernel<384, 0><<<GB, 256, 0, stream>>>(buf0, wq + 256 * 128, qkv_b, buf1,
                                                 nullptr, nullptr, nullptr, 256);

    // 4. shifted-window attention -> buf0
    attn_kernel<<<TOTW / 8, 256, 0, stream>>>(buf1, rel_b, buf0);

    // 5. proj GEMM + residual(x transposed) -> xo f32
    gemm2_kernel<128, 2><<<GB, 256, 0, stream>>>(buf0, wp, proj_b, nullptr,
                                                 buf2, x, nullptr, 0);

    // 6. LN2 -> bf16
    ln2_kernel<<<M / 8, 256, 0, stream>>>(buf2, n2_w, n2_b, buf0);

    // 7. fc1 GEMM + exact GELU -> bf16
    gemm2_kernel<128, 1><<<GB, 256, 0, stream>>>(buf0, w1, fc1_b, buf1,
                                                 nullptr, nullptr, nullptr, 0);

    // 8. fc2 GEMM + residual(xo) -> d_out in (B,T,C,H,W)
    gemm2_kernel<128, 3><<<GB, 256, 0, stream>>>(buf1, w2, fc2_b, nullptr,
                                                 out, nullptr, buf2, 0);
}
